// SoilMoistureGapFilling_52218212384939
// MI455X (gfx1250) — matrix operand order repaired and checked by reading, not compiled, not run on hardware
//
#include <hip/hip_runtime.h>

// ---------------------------------------------------------------------------
// SoilMoistureGapFilling: 365-step LSTM (B=1024, H=128) + FFN head, on CDNA5.
// Weights staged in LDS as f16, recurrent matmul via v_wmma_f32_16x16x32_f16.
// ---------------------------------------------------------------------------

typedef __attribute__((ext_vector_type(16))) _Float16 v16h;
typedef __attribute__((ext_vector_type(8)))  float    v8f;

constexpr int kB   = 1024;
constexpr int kT   = 365;
constexpr int kF   = 8;
constexpr int kS   = 16;
constexpr int kH   = 128;
constexpr int kFFN = 128;
constexpr int kG   = 4 * kH;    // 512 gate rows
constexpr int kKE  = 160;       // extended K: 128 (h) + 32 (inp/static, zero padded)
constexpr int kTB  = 16;        // batch rows per workgroup

// ---------------------------------------------------------------------------
// Prep: pack f16 extended weight matrices + bias sum into workspace.
//   wext  [512][160] f16 : cols 0..127 = W_hh, 128..136 = W_ih, rest 0
//   w1ext [128][160] f16 : cols 0..143 = W1, rest 0
//   bias_sum[512] f32    : b_ih + b_hh
// ---------------------------------------------------------------------------
__global__ void prep_kernel(const float* __restrict__ W_ih,
                            const float* __restrict__ W_hh,
                            const float* __restrict__ b_ih,
                            const float* __restrict__ b_hh,
                            const float* __restrict__ W1,
                            _Float16* __restrict__ wext,
                            _Float16* __restrict__ w1ext,
                            float* __restrict__ bias_sum) {
  int idx = blockIdx.x * blockDim.x + threadIdx.x;
  const int NW = kG * kKE;           // 81920
  const int NW1 = kFFN * kKE;        // 20480
  if (idx < NW) {
    int g = idx / kKE, k = idx % kKE;
    float v = 0.0f;
    if (k < kH)              v = W_hh[g * kH + k];
    else if (k < kH + 1 + kF) v = W_ih[g * (kF + 1) + (k - kH)];
    wext[idx] = (_Float16)v;
  } else if (idx < NW + NW1) {
    int j = (idx - NW) / kKE, k = (idx - NW) % kKE;
    float v = (k < kH + kS) ? W1[j * (kH + kS) + k] : 0.0f;
    w1ext[idx - NW] = (_Float16)v;
  } else if (idx < NW + NW1 + kG) {
    int g = idx - NW - NW1;
    bias_sum[g] = b_ih[g] + b_hh[g];
  }
}

// Load a 16x32 f16 A-operand tile slice for this lane from an LDS matrix with
// row pitch kKE halves.  Layout: lane<16 holds row M=lane, K {cb..cb+15};
// lane>=16 holds row M=lane-16 with cb offset by 16.
__device__ __forceinline__ v16h load_a(const _Float16* Wl, int row, int cb) {
  const _Float16* p = Wl + row * kKE + cb;
  v16h a;
  *((uint4*)&a)     = *(const uint4*)p;         // 8 halves, 16B aligned
  *(((uint4*)&a)+1) = *(const uint4*)(p + 8);   // next 8 halves
  return a;
}

__device__ __forceinline__ float sigf(float x) {
  return 1.0f / (1.0f + __expf(-x));
}

__global__ __launch_bounds__(256, 1)
void lstm_main(const float* __restrict__ x,
               const float* __restrict__ tvf,     // (B,T,F)
               const float* __restrict__ statf,   // (B,T,S)
               const float* __restrict__ b1,
               const float* __restrict__ W2,
               const float* __restrict__ b2,
               const _Float16* __restrict__ wext_g,   // contiguous: wext|w1ext
               const float* __restrict__ bias_sum,
               float* __restrict__ out) {
  extern __shared__ char smem[];
  _Float16* hW   = (_Float16*)smem;              // 512*160 f16
  _Float16* hW1  = hW  + kG * kKE;               // 128*160 f16
  _Float16* hbuf = hW1 + kFFN * kKE;             // h state: [16 batch][128] f16
  float* y_cur = (float*)(hbuf + kTB * kH);      // 16 f32
  float* yacc  = y_cur + kTB;                    // 16 f32

  const int tid  = threadIdx.x;
  const int w    = tid >> 5;         // wave 0..7
  const int lane = tid & 31;
  const int hi   = (lane >> 4) & 1;  // lane group
  const int ln   = lane & 15;        // batch column within tile
  const int bbase = blockIdx.x * kTB;

  // ---- one-time cooperative copy of f16 weights into LDS ----
  {
    const uint4* src = (const uint4*)wext_g;
    uint4* dst = (uint4*)hW;
    const int n16 = (kG + kFFN) * kKE * 2 / 16;  // 12800 uint4
    for (int i = tid; i < n16; i += blockDim.x) dst[i] = src[i];
  }
  // zero h state, init y_cur = x[:,0,0], yacc = 0
  {
    uint4 z{0u, 0u, 0u, 0u};
    uint4* hz = (uint4*)hbuf;
    for (int i = tid; i < kTB * kH * 2 / 16; i += blockDim.x) hz[i] = z;
    if (tid < kTB) {
      y_cur[tid] = x[(size_t)(bbase + tid) * kT];
      yacc[tid]  = 0.0f;
    }
  }

  // ---- per-wave/lane constants (registers, loaded once) ----
  float biasr[4][8], b1r[8], w2r[8];
#pragma unroll
  for (int q = 0; q < 4; ++q)
#pragma unroll
    for (int r = 0; r < 8; ++r)
      biasr[q][r] = bias_sum[(w + 8 * q) * 16 + r + hi * 8];
#pragma unroll
  for (int r = 0; r < 8; ++r) {
    b1r[r] = b1[w * 16 + r + hi * 8];
    w2r[r] = W2[w * 16 + r + hi * 8];
  }
  const float b2s = b2[0];
  float creg[8];
#pragma unroll
  for (int r = 0; r < 8; ++r) creg[r] = 0.0f;

  __syncthreads();

  for (int t = 0; t < kT; ++t) {
    // ---- per-step streaming inputs ----
    float yv = y_cur[ln];
    v16h binp, bstat;
#pragma unroll
    for (int i = 0; i < 16; ++i) { binp[i] = (_Float16)0; bstat[i] = (_Float16)0; }
    if (!hi) {
      const float4* fp =
          (const float4*)(tvf + ((size_t)(bbase + ln) * kT + t) * kF);
      float4 f0 = fp[0], f1 = fp[1];
      binp[0] = (_Float16)yv;
      binp[1] = (_Float16)f0.x; binp[2] = (_Float16)f0.y;
      binp[3] = (_Float16)f0.z; binp[4] = (_Float16)f0.w;
      binp[5] = (_Float16)f1.x; binp[6] = (_Float16)f1.y;
      binp[7] = (_Float16)f1.z; binp[8] = (_Float16)f1.w;
      const float4* sp =
          (const float4*)(statf + ((size_t)(bbase + ln) * kT + t) * kS);
      float4 s0 = sp[0], s1 = sp[1], s2 = sp[2], s3 = sp[3];
      bstat[0]  = (_Float16)s0.x; bstat[1]  = (_Float16)s0.y;
      bstat[2]  = (_Float16)s0.z; bstat[3]  = (_Float16)s0.w;
      bstat[4]  = (_Float16)s1.x; bstat[5]  = (_Float16)s1.y;
      bstat[6]  = (_Float16)s1.z; bstat[7]  = (_Float16)s1.w;
      bstat[8]  = (_Float16)s2.x; bstat[9]  = (_Float16)s2.y;
      bstat[10] = (_Float16)s2.z; bstat[11] = (_Float16)s2.w;
      bstat[12] = (_Float16)s3.x; bstat[13] = (_Float16)s3.y;
      bstat[14] = (_Float16)s3.z; bstat[15] = (_Float16)s3.w;
    }

    // ---- gates^T = W_ext(512x160) x [h; inp]^T(160x16) ----
    v8f acc[4];
#pragma unroll
    for (int q = 0; q < 4; ++q) acc[q] = (v8f){};
#pragma unroll
    for (int kt = 0; kt < 5; ++kt) {
      v16h bm;
      if (kt < 4) {
        const _Float16* p = hbuf + ln * kH + kt * 32 + hi * 16;
        *((uint4*)&bm)     = *(const uint4*)p;
        *(((uint4*)&bm)+1) = *(const uint4*)(p + 8);
      } else {
        bm = binp;
      }
      const int cb = kt * 32 + hi * 16;
#pragma unroll
      for (int q = 0; q < 4; ++q) {
        v16h a = load_a(hW, (w + 8 * q) * 16 + ln, cb);
        acc[q] = __builtin_amdgcn_wmma_f32_16x16x32_f16(
            false, a, false, bm, (short)0, acc[q], false, false);
      }
    }
    __syncthreads();   // all reads of old h / y_cur done

    // ---- LSTM pointwise: wave w owns hidden dims [16w,16w+16) ----
#pragma unroll
    for (int r = 0; r < 8; ++r) {
      float xi = acc[0][r] + biasr[0][r];
      float xf = acc[1][r] + biasr[1][r];
      float xg = acc[2][r] + biasr[2][r];
      float xo = acc[3][r] + biasr[3][r];
      float i_ = sigf(xi);
      float f_ = sigf(xf);
      float g_ = tanhf(xg);
      float o_ = sigf(xo);
      float c  = f_ * creg[r] + i_ * g_;
      creg[r]  = c;
      float hv = o_ * tanhf(c);
      hbuf[ln * kH + w * 16 + hi * 8 + r] = (_Float16)hv;
    }
    __syncthreads();   // new h visible

    // ---- FFN: hid^T = W1_ext(128x160) x [h; stat]^T ; wave w -> tile w ----
    v8f accF = (v8f){};
#pragma unroll
    for (int kt = 0; kt < 5; ++kt) {
      v16h bm;
      if (kt < 4) {
        const _Float16* p = hbuf + ln * kH + kt * 32 + hi * 16;
        *((uint4*)&bm)     = *(const uint4*)p;
        *(((uint4*)&bm)+1) = *(const uint4*)(p + 8);
      } else {
        bm = bstat;
      }
      v16h a = load_a(hW1, w * 16 + ln, kt * 32 + hi * 16);
      accF = __builtin_amdgcn_wmma_f32_16x16x32_f16(
          false, a, false, bm, (short)0, accF, false, false);
    }
    // relu + dot with W2 (per-lane partial over its 8 FFN dims)
    float part = 0.0f;
#pragma unroll
    for (int r = 0; r < 8; ++r)
      part += fmaxf(accF[r] + b1r[r], 0.0f) * w2r[r];
    atomicAdd(&yacc[ln], part);       // ds_add_f32
    __syncthreads();

    if (w == 0 && lane < kTB) {
      float y = yacc[lane] + b2s;
      y_cur[lane] = y;
      out[(size_t)(bbase + lane) * kT + t] = y;
      yacc[lane] = 0.0f;
    }
    __syncthreads();
  }
}

extern "C" void kernel_launch(void* const* d_in, const int* in_sizes, int n_in,
                              void* d_out, int out_size, void* d_ws, size_t ws_size,
                              hipStream_t stream) {
  const float* x     = (const float*)d_in[0];
  // d_in[1] = mask (unused by reference math)
  const float* tvf   = (const float*)d_in[2];
  const float* statf = (const float*)d_in[3];
  const float* W_ih  = (const float*)d_in[4];
  const float* W_hh  = (const float*)d_in[5];
  const float* b_ih  = (const float*)d_in[6];
  const float* b_hh  = (const float*)d_in[7];
  const float* W1    = (const float*)d_in[8];
  const float* b1    = (const float*)d_in[9];
  const float* W2    = (const float*)d_in[10];
  const float* b2    = (const float*)d_in[11];
  float* out = (float*)d_out;

  // workspace layout: wext f16 | w1ext f16 | bias_sum f32
  _Float16* wext  = (_Float16*)d_ws;
  _Float16* w1ext = wext + kG * kKE;
  float* bias_sum = (float*)(w1ext + kFFN * kKE);

  const int prepN = kG * kKE + kFFN * kKE + kG;
  prep_kernel<<<(prepN + 255) / 256, 256, 0, stream>>>(
      W_ih, W_hh, b_ih, b_hh, W1, wext, w1ext, bias_sum);

  const size_t shmem = (size_t)(kG * kKE + kFFN * kKE + kTB * kH) * sizeof(_Float16)
                     + 2 * kTB * sizeof(float);
  hipFuncSetAttribute(reinterpret_cast<const void*>(lstm_main),
                      hipFuncAttributeMaxDynamicSharedMemorySize, (int)shmem);

  lstm_main<<<kB / kTB, 256, shmem, stream>>>(
      x, tvf, statf, b1, W2, b2, wext, bias_sum, out);
}
